// GeneGAT_71373766525395
// MI455X (gfx1250) — matrix operand order repaired and checked by reading, not compiled, not run on hardware
//
#include <hip/hip_runtime.h>
#include <hip/hip_bf16.h>
#include <limits.h>

// ---------------------------------------------------------------------------
// GATv2 x2 pipeline for gfx1250 (wave32).  Fixed dims from the reference:
//   D = 64, H1*C1 = 1024, fused GEMM N-dim = 3072 (xl1|xr1|xskip)
// ---------------------------------------------------------------------------

#define D_IN      64
#define H1        8
#define C1        128
#define F1        1024          // H1*C1
#define NCAT      3072          // xl | xr | xskip
#define NBLK      24            // NCAT / 128 (N-cols per workgroup)
#define LRELU     0.2f

typedef __attribute__((ext_vector_type(16))) _Float16 v16h;
typedef __attribute__((ext_vector_type(8)))  float    v8f;
typedef __attribute__((ext_vector_type(4)))  int      v4i;

// address-space-qualified v4i for the async-copy builtin
typedef __attribute__((address_space(1))) v4i gv4i;   // global
typedef __attribute__((address_space(3))) v4i lv4i;   // LDS

#if __has_builtin(__builtin_amdgcn_global_load_async_to_lds_b128)
#define HAVE_ASYNC_LDS 1
#else
#define HAVE_ASYNC_LDS 0
#endif

__device__ __forceinline__ void wait_asynccnt0() {
#if __has_builtin(__builtin_amdgcn_s_wait_asynccnt)
    __builtin_amdgcn_s_wait_asynccnt(0);
#else
    asm volatile("s_wait_asynccnt 0x0" ::: "memory");
#endif
}

// ---- helpers ---------------------------------------------------------------

// order-preserving float <-> int map so we can use integer atomicMax
__device__ __forceinline__ int ford(float f) {
    int i = __float_as_int(f);
    return (i < 0) ? (i ^ 0x7fffffff) : i;
}
__device__ __forceinline__ float dford(int i) {
    return __int_as_float((i < 0) ? (i ^ 0x7fffffff) : i);
}

__device__ __forceinline__ float wred32(float v) {
    v += __shfl_xor(v, 16, 32);
    v += __shfl_xor(v,  8, 32);
    v += __shfl_xor(v,  4, 32);
    v += __shfl_xor(v,  2, 32);
    v += __shfl_xor(v,  1, 32);
    return v;
}

__device__ __forceinline__ void edge_nodes(const long long* __restrict__ ei,
                                           long long e, long long E,
                                           int& s, int& d) {
    if (e < E) { s = (int)ei[e]; d = (int)ei[E + e]; }
    else       { int n = (int)(e - E); s = n; d = n; }   // self loops
}

__device__ __forceinline__ float lrelu_f(float v) { return v > 0.f ? v : LRELU * v; }

// ---- init / prep kernels ---------------------------------------------------

__global__ void fill_f32(float* __restrict__ p, float v, long long n) {
    long long i = (long long)blockIdx.x * blockDim.x + threadIdx.x;
    long long stride = (long long)gridDim.x * blockDim.x;
    for (; i < n; i += stride) p[i] = v;
}

__global__ void fill_i32(int* __restrict__ p, int v, long long n) {
    long long i = (long long)blockIdx.x * blockDim.x + threadIdx.x;
    long long stride = (long long)gridDim.x * blockDim.x;
    for (; i < n; i += stride) p[i] = v;
}

__global__ void init_out2(float* __restrict__ out, const float* __restrict__ b2, int n) {
    int i = blockIdx.x * blockDim.x + threadIdx.x;
    if (i < n) out[i] = b2[i & 1];
}

__global__ void cvt_x_f16(const float* __restrict__ x, _Float16* __restrict__ xh, long long n) {
    long long i = (long long)blockIdx.x * blockDim.x + threadIdx.x;
    if (i < n) xh[i] = (_Float16)x[i];
}

// wt[n*64+k] = W(k, c) with (W,c) selected by output column n  -> N-major f16
__global__ void pack_w1(const float* __restrict__ Wl1, const float* __restrict__ Wr1,
                        const float* __restrict__ Wsk, _Float16* __restrict__ wt) {
    int i = blockIdx.x * blockDim.x + threadIdx.x;
    if (i >= NCAT * D_IN) return;
    int n = i >> 6, k = i & 63;
    const float* W = (n < F1) ? Wl1 : ((n < 2 * F1) ? Wr1 : Wsk);
    int c = n & (F1 - 1);
    wt[i] = (_Float16)W[k * F1 + c];
}

// ---- fused GEMM:  XLRS[N,3072] = x(f16) @ [Wl1|Wr1|Wskip](f16) + bias ------
// Workgroup = 16 M-rows x 128 N-cols.  A panel (2 KB) + B panel (16 KB) are
// staged into LDS with GLOBAL_LOAD_ASYNC_TO_LDS_B128 (ASYNCcnt), then each of
// the 8 waves runs two v_wmma_f32_16x16x32_f16 steps on its 16x16 tile.
__global__ __launch_bounds__(256)
void gemm1_wmma(const _Float16* __restrict__ A,   // [N,64] row-major
                const _Float16* __restrict__ Bt,  // [3072,64] (N-major)
                const float* __restrict__ bl1, const float* __restrict__ br1,
                const float* __restrict__ bsk,
                float* __restrict__ C) {
    __shared__ __align__(16) _Float16 lA[16 * D_IN];    //  2 KB
    __shared__ __align__(16) _Float16 lB[128 * D_IN];   // 16 KB

    int tid  = threadIdx.x;
    int mt   = blockIdx.x / NBLK;           // M tile (16 rows)
    int nbk  = blockIdx.x % NBLK;           // 128-column block

    const _Float16* gA = A  + (size_t)mt  * 16  * D_IN;
    const _Float16* gB = Bt + (size_t)nbk * 128 * D_IN;

#if HAVE_ASYNC_LDS
    if (tid < 128)                                    // A: 128 x 16B
        __builtin_amdgcn_global_load_async_to_lds_b128(
            (gv4i*)(gA + tid * 8), (lv4i*)(lA + tid * 8), 0, 0);
#pragma unroll
    for (int j = 0; j < 4; ++j) {                     // B: 1024 x 16B
        int c = tid + j * 256;
        __builtin_amdgcn_global_load_async_to_lds_b128(
            (gv4i*)(gB + c * 8), (lv4i*)(lB + c * 8), 0, 0);
    }
    wait_asynccnt0();
#else
    if (tid < 128) *(v4i*)(lA + tid * 8) = *(const v4i*)(gA + tid * 8);
#pragma unroll
    for (int j = 0; j < 4; ++j) {
        int c = tid + j * 256;
        *(v4i*)(lB + c * 8) = *(const v4i*)(gB + c * 8);
    }
#endif
    __syncthreads();

    int wv   = tid >> 5;                    // wave -> local N tile (0..7)
    int lane = tid & 31;
    int half = lane >> 4;
    int l16  = lane & 15;

    const _Float16* arow = lA + l16 * D_IN;                 // M = l16
    const _Float16* brow = lB + (wv * 16 + l16) * D_IN;     // Ncol = l16

    v8f c = {};
#pragma unroll
    for (int ks = 0; ks < 2; ++ks) {
        int kb = ks * 32;
        union { v4i q[2]; v16h h; } au, bu;
        // A 16x32 f16 fragment: half==0 -> K {0..7,16..23}; half==1 -> {8..15,24..31}
        au.q[0] = *(const v4i*)(arow + kb + half * 8);
        au.q[1] = *(const v4i*)(arow + kb + half * 8 + 16);
        // B 32x16 f16 fragment: half==0 -> K {0..7,16..23}; half==1 -> {8..15,24..31}
        bu.q[0] = *(const v4i*)(brow + kb + half * 8);
        bu.q[1] = *(const v4i*)(brow + kb + half * 8 + 16);
        c = __builtin_amdgcn_wmma_f32_16x16x32_f16(false, au.h, false, bu.h,
                                                   (short)0, c, false, false);
    }

    int col = nbk * 128 + wv * 16 + l16;
    float bias = (col < F1) ? bl1[col]
               : (col < 2 * F1) ? br1[col - F1] : bsk[col - 2 * F1];
#pragma unroll
    for (int i = 0; i < 8; ++i) {
        int row = mt * 16 + half * 8 + i;   // C layout: lanes0-15 M=i, lanes16-31 M=i+8
        C[(size_t)row * NCAT + col] = c[i] + bias;
    }
}

// ---- conv1 edge attention ---------------------------------------------------

// one wave per edge: logits alpha1[e,h] + segment max over dst
__global__ __launch_bounds__(256)
void elog1(const float* __restrict__ XLRS, const long long* __restrict__ ei,
           const float* __restrict__ att1, float* __restrict__ alpha1,
           int* __restrict__ amax1, long long E, long long E2) {
    long long wave = ((long long)blockIdx.x * blockDim.x + threadIdx.x) >> 5;
    int lane = threadIdx.x & 31;
    if (wave >= E2) return;
    int s, d; edge_nodes(ei, wave, E, s, d);
    const float* xl = XLRS + (size_t)s * NCAT;        // cols [0,1024)
    const float* xr = XLRS + (size_t)d * NCAT + F1;   // cols [1024,2048)
    float p = 0.f;
    int base = lane * 32;                              // head = lane/4 (128ch/head)
#pragma unroll
    for (int j = 0; j < 8; ++j) {
        int idx = base + j * 4;
        float4 a = *(const float4*)(xl + idx);
        float4 b = *(const float4*)(xr + idx);
        float4 t = *(const float4*)(att1 + idx);       // att1 flat = h*128+c
        p += t.x * lrelu_f(a.x + b.x);
        p += t.y * lrelu_f(a.y + b.y);
        p += t.z * lrelu_f(a.z + b.z);
        p += t.w * lrelu_f(a.w + b.w);
    }
    p += __shfl_xor(p, 1, 32);
    p += __shfl_xor(p, 2, 32);                         // sum within 4-lane head group
    if ((lane & 3) == 0) {
        int h = lane >> 2;
        alpha1[wave * 8 + h] = p;
        atomicMax(&amax1[(size_t)d * 8 + h], ford(p));
    }
}

// exp(alpha - max) and denominators; one thread per (edge, head)
__global__ void soft1(float* __restrict__ alpha1, const int* __restrict__ amax1,
                      float* __restrict__ denom1, const long long* __restrict__ ei,
                      long long E, long long E2) {
    long long tid = (long long)blockIdx.x * blockDim.x + threadIdx.x;
    if (tid >= E2 * 8) return;
    long long e = tid >> 3; int h = (int)(tid & 7);
    int s, d; edge_nodes(ei, e, E, s, d); (void)s;
    float ex = __expf(alpha1[tid] - dford(amax1[(size_t)d * 8 + h]));
    alpha1[tid] = ex;
    atomicAdd(&denom1[(size_t)d * 8 + h], ex);
}

// weighted scatter-sum of messages: out1[dst] += xl1[src] * alpha; wave per edge
__global__ __launch_bounds__(256)
void scatter1(const float* __restrict__ XLRS, const float* __restrict__ alpha1,
              const float* __restrict__ denom1, float* __restrict__ out1,
              const long long* __restrict__ ei, long long E, long long E2) {
    long long wave = ((long long)blockIdx.x * blockDim.x + threadIdx.x) >> 5;
    int lane = threadIdx.x & 31;
    if (wave >= E2) return;
    int s, d; edge_nodes(ei, wave, E, s, d);
    int h = lane >> 2;
    float w = alpha1[wave * 8 + h] / (denom1[(size_t)d * 8 + h] + 1e-16f);
    const float* xl = XLRS + (size_t)s * NCAT;
    float* orow = out1 + (size_t)d * F1;
    int base = lane * 32;
#pragma unroll
    for (int j = 0; j < 8; ++j) {
        int idx = base + j * 4;
        float4 a = *(const float4*)(xl + idx);
        atomicAdd(orow + idx + 0, a.x * w);
        atomicAdd(orow + idx + 1, a.y * w);
        atomicAdd(orow + idx + 2, a.z * w);
        atomicAdd(orow + idx + 3, a.w * w);
    }
}

// x1 = ELU(LayerNorm(out1 + b1 + xskip)); in-place over out1. One wave per node.
__global__ __launch_bounds__(256)
void fuse_ln(float* __restrict__ x1, const float* __restrict__ XLRS,
             const float* __restrict__ b1, const float* __restrict__ g1,
             const float* __restrict__ beta1, int N) {
    int node = (int)(((long long)blockIdx.x * blockDim.x + threadIdx.x) >> 5);
    int lane = threadIdx.x & 31;
    if (node >= N) return;
    float* row = x1 + (size_t)node * F1;
    const float* xs = XLRS + (size_t)node * NCAT + 2 * F1;
    float4 r[8];
    float s = 0.f, s2 = 0.f;
#pragma unroll
    for (int j = 0; j < 8; ++j) {
        int idx = j * 128 + lane * 4;
        float4 v  = *(const float4*)(row + idx);
        float4 sk = *(const float4*)(xs + idx);
        float4 bb = *(const float4*)(b1 + idx);
        v.x += sk.x + bb.x; v.y += sk.y + bb.y;
        v.z += sk.z + bb.z; v.w += sk.w + bb.w;
        r[j] = v;
        s  += v.x + v.y + v.z + v.w;
        s2 += v.x * v.x + v.y * v.y + v.z * v.z + v.w * v.w;
    }
    s = wred32(s); s2 = wred32(s2);
    float mu  = s * (1.f / F1);
    float var = s2 * (1.f / F1) - mu * mu;
    float rs  = rsqrtf(var + 1e-5f);
#pragma unroll
    for (int j = 0; j < 8; ++j) {
        int idx = j * 128 + lane * 4;
        float4 g  = *(const float4*)(g1 + idx);
        float4 bt = *(const float4*)(beta1 + idx);
        float4 v  = r[j];
        float y0 = (v.x - mu) * rs * g.x + bt.x;
        float y1 = (v.y - mu) * rs * g.y + bt.y;
        float y2 = (v.z - mu) * rs * g.z + bt.z;
        float y3 = (v.w - mu) * rs * g.w + bt.w;
        v.x = y0 > 0.f ? y0 : __expf(y0) - 1.f;
        v.y = y1 > 0.f ? y1 : __expf(y1) - 1.f;
        v.z = y2 > 0.f ? y2 : __expf(y2) - 1.f;
        v.w = y3 > 0.f ? y3 : __expf(y3) - 1.f;
        *(float4*)(row + idx) = v;
    }
}

// ---- conv2 (1024 -> 2, heads=1) --------------------------------------------

// xlr2[n] = {xl2_0, xl2_1, xr2_0, xr2_1}; one wave per node
__global__ __launch_bounds__(256)
void lin2(const float* __restrict__ x1, const float* __restrict__ Wl2,
          const float* __restrict__ bl2, const float* __restrict__ Wr2,
          const float* __restrict__ br2, float* __restrict__ xlr2, int N) {
    int node = (int)(((long long)blockIdx.x * blockDim.x + threadIdx.x) >> 5);
    int lane = threadIdx.x & 31;
    if (node >= N) return;
    const float* row = x1 + (size_t)node * F1;
    float a0 = 0.f, a1 = 0.f, c0 = 0.f, c1 = 0.f;
#pragma unroll 8
    for (int j = 0; j < 32; ++j) {
        int k = j * 32 + lane;
        float v = row[k];
        a0 += v * Wl2[k * 2 + 0]; a1 += v * Wl2[k * 2 + 1];
        c0 += v * Wr2[k * 2 + 0]; c1 += v * Wr2[k * 2 + 1];
    }
    a0 = wred32(a0); a1 = wred32(a1); c0 = wred32(c0); c1 = wred32(c1);
    if (lane == 0) {
        xlr2[(size_t)node * 4 + 0] = a0 + bl2[0];
        xlr2[(size_t)node * 4 + 1] = a1 + bl2[1];
        xlr2[(size_t)node * 4 + 2] = c0 + br2[0];
        xlr2[(size_t)node * 4 + 3] = c1 + br2[1];
    }
}

__global__ void elog2(const float* __restrict__ xlr2, const long long* __restrict__ ei,
                      const float* __restrict__ att2, float* __restrict__ alpha2,
                      int* __restrict__ amax2, long long E, long long E2) {
    long long e = (long long)blockIdx.x * blockDim.x + threadIdx.x;
    if (e >= E2) return;
    int s, d; edge_nodes(ei, e, E, s, d);
    float e0 = lrelu_f(xlr2[(size_t)s * 4 + 0] + xlr2[(size_t)d * 4 + 2]);
    float e1 = lrelu_f(xlr2[(size_t)s * 4 + 1] + xlr2[(size_t)d * 4 + 3]);
    float lg = att2[0] * e0 + att2[1] * e1;
    alpha2[e] = lg;
    atomicMax(&amax2[d], ford(lg));
}

__global__ void soft2(float* __restrict__ alpha2, const int* __restrict__ amax2,
                      float* __restrict__ denom2, const long long* __restrict__ ei,
                      long long E, long long E2) {
    long long e = (long long)blockIdx.x * blockDim.x + threadIdx.x;
    if (e >= E2) return;
    int s, d; edge_nodes(ei, e, E, s, d); (void)s;
    float ex = __expf(alpha2[e] - dford(amax2[d]));
    alpha2[e] = ex;
    atomicAdd(&denom2[d], ex);
}

__global__ void scatter2(const float* __restrict__ xlr2, const float* __restrict__ alpha2,
                         const float* __restrict__ denom2, float* __restrict__ out,
                         const long long* __restrict__ ei, long long E, long long E2) {
    long long e = (long long)blockIdx.x * blockDim.x + threadIdx.x;
    if (e >= E2) return;
    int s, d; edge_nodes(ei, e, E, s, d);
    float w = alpha2[e] / (denom2[d] + 1e-16f);
    atomicAdd(&out[(size_t)d * 2 + 0], xlr2[(size_t)s * 4 + 0] * w);
    atomicAdd(&out[(size_t)d * 2 + 1], xlr2[(size_t)s * 4 + 1] * w);
}

// ---------------------------------------------------------------------------

extern "C" void kernel_launch(void* const* d_in, const int* in_sizes, int n_in,
                              void* d_out, int out_size, void* d_ws, size_t ws_size,
                              hipStream_t stream) {
    const float*     x     = (const float*)d_in[0];
    const long long* ei    = (const long long*)d_in[1];   // int64 edge_index [2,E]
    const float*     Wl1   = (const float*)d_in[2];
    const float*     bl1   = (const float*)d_in[3];
    const float*     Wr1   = (const float*)d_in[4];
    const float*     br1   = (const float*)d_in[5];
    const float*     att1  = (const float*)d_in[6];
    const float*     b1    = (const float*)d_in[7];
    const float*     Wsk   = (const float*)d_in[8];
    const float*     bsk   = (const float*)d_in[9];
    const float*     g1    = (const float*)d_in[10];
    const float*     beta1 = (const float*)d_in[11];
    const float*     Wl2   = (const float*)d_in[12];
    const float*     bl2   = (const float*)d_in[13];
    const float*     Wr2   = (const float*)d_in[14];
    const float*     br2   = (const float*)d_in[15];
    const float*     att2  = (const float*)d_in[16];
    const float*     b2    = (const float*)d_in[17];
    float* out = (float*)d_out;

    const int       N  = in_sizes[0] / D_IN;       // 20000 (multiple of 16)
    const long long E  = in_sizes[1] / 2;          // 160000
    const long long E2 = E + N;                    // with self loops
    const int mtiles = N / 16;

    // ---- workspace carve ----------------------------------------------------
    char* ws = (char*)d_ws;
    _Float16* xh   = (_Float16*)ws; ws += (size_t)N * D_IN * sizeof(_Float16);
    _Float16* wt1  = (_Float16*)ws; ws += (size_t)NCAT * D_IN * sizeof(_Float16);
    float* XLRS    = (float*)ws;    ws += (size_t)N * NCAT * sizeof(float);
    float* out1    = (float*)ws;    ws += (size_t)N * F1 * sizeof(float);   // later x1
    float* alpha1  = (float*)ws;    ws += (size_t)E2 * 8 * sizeof(float);
    float* denom1  = (float*)ws;    ws += (size_t)N * 8 * sizeof(float);
    int*   amax1   = (int*)ws;      ws += (size_t)N * 8 * sizeof(int);
    float* xlr2    = (float*)ws;    ws += (size_t)N * 4 * sizeof(float);
    float* alpha2  = (float*)ws;    ws += (size_t)E2 * sizeof(float);
    float* denom2  = (float*)ws;    ws += (size_t)N * sizeof(float);
    int*   amax2   = (int*)ws;      ws += (size_t)N * sizeof(int);
    (void)ws_size; (void)n_in;

    const int T = 256;
    auto nb = [](long long n, int t) { return (int)((n + t - 1) / t); };

    // ---- re-init accumulators every call (harness does not re-poison) ------
    fill_f32<<<2048, T, 0, stream>>>(out1, 0.f, (long long)N * F1);
    fill_f32<<<64,   T, 0, stream>>>(denom1, 0.f, (long long)N * 8);
    fill_i32<<<64,   T, 0, stream>>>(amax1, INT_MIN, (long long)N * 8);
    fill_f32<<<32,   T, 0, stream>>>(denom2, 0.f, (long long)N);
    fill_i32<<<32,   T, 0, stream>>>(amax2, INT_MIN, (long long)N);
    init_out2<<<nb(out_size, T), T, 0, stream>>>(out, b2, out_size);

    // ---- prep: f16 activations + N-major f16 fused weight ------------------
    cvt_x_f16<<<nb((long long)N * D_IN, T), T, 0, stream>>>(x, xh, (long long)N * D_IN);
    pack_w1<<<nb(NCAT * D_IN, T), T, 0, stream>>>(Wl1, Wr1, Wsk, wt1);

    // ---- fused WMMA GEMM: XLRS = x @ [Wl1|Wr1|Wskip] + bias ----------------
    gemm1_wmma<<<mtiles * NBLK, T, 0, stream>>>(xh, wt1, bl1, br1, bsk, XLRS);

    // ---- conv1 attention + aggregate ---------------------------------------
    elog1<<<nb(E2 * 32, T), T, 0, stream>>>(XLRS, ei, att1, alpha1, amax1, E, E2);
    soft1<<<nb(E2 * 8, T), T, 0, stream>>>(alpha1, amax1, denom1, ei, E, E2);
    scatter1<<<nb(E2 * 32, T), T, 0, stream>>>(XLRS, alpha1, denom1, out1, ei, E, E2);

    // ---- bias + skip + LayerNorm + ELU (in place -> x1) --------------------
    fuse_ln<<<nb((long long)N * 32, T), T, 0, stream>>>(out1, XLRS, b1, g1, beta1, N);

    // ---- conv2 --------------------------------------------------------------
    lin2<<<nb((long long)N * 32, T), T, 0, stream>>>(out1, Wl2, bl2, Wr2, br2, xlr2, N);
    elog2<<<nb(E2, T), T, 0, stream>>>(xlr2, ei, att2, alpha2, amax2, E, E2);
    soft2<<<nb(E2, T), T, 0, stream>>>(alpha2, amax2, denom2, ei, E, E2);
    scatter2<<<nb(E2, T), T, 0, stream>>>(xlr2, alpha2, denom2, out, ei, E, E2);
}
